// TCC_Conv_56487409877746
// MI455X (gfx1250) — compile-verified, matrix operand order checked by reading, not run on hardware
//
#include <hip/hip_runtime.h>
#include <math.h>

#define N_B    4
#define C_IN   256
#define NC     32
#define TR     8
#define NK     3
#define HH     128
#define WW     128
#define HW     (HH*WW)
#define K_OFF  (NC*9)   /* 288 */
#define OFF_CH 27

typedef __bf16 bf16_t;
typedef __attribute__((ext_vector_type(16))) __bf16 bfx16;
typedef __attribute__((ext_vector_type(8)))  float  fx8;

__device__ __forceinline__ int kmap(int e, int hi) {
  // ISA 16-bit A/B layout: lane half 'hi' selects K sub-range
  return (e < 8 ? e : e + 8) + 8 * hi;
}
__device__ __forceinline__ fx8 wmma_bf16(bfx16 a, bfx16 b, fx8 c) {
  return __builtin_amdgcn_wmma_f32_16x16x32_bf16(false, a, false, b, (short)0, c,
                                                 false, false);
}
__device__ __forceinline__ float sigmoidf_(float x) { return 1.f / (1.f + expf(-x)); }

// ---------------------------------------------------------------- weight staging (fp32 -> bf16, pad M/K)
__global__ void k_stage(const float* __restrict__ in_w, const float* __restrict__ qoff,
                        const float* __restrict__ koff, const float* __restrict__ ow,
                        bf16_t* __restrict__ w_in, bf16_t* __restrict__ w_q,
                        bf16_t* __restrict__ w_k, bf16_t* __restrict__ w_o) {
  int t = blockIdx.x * blockDim.x + threadIdx.x;
  if (t < NC * C_IN) { w_in[t] = (bf16_t)in_w[t]; return; }
  t -= NC * C_IN;
  if (t < 32 * K_OFF) { int m = t / K_OFF; w_q[t] = (bf16_t)(m < OFF_CH ? qoff[t] : 0.f); return; }
  t -= 32 * K_OFF;
  if (t < 32 * K_OFF) { int m = t / K_OFF; w_k[t] = (bf16_t)(m < OFF_CH ? koff[t] : 0.f); return; }
  t -= 32 * K_OFF;
  if (t < C_IN * 32) {
    int o = t / 32, j = t % 32;
    w_o[t] = (bf16_t)(j < OFF_CH ? ow[o * OFF_CH + j] : 0.f);
  }
}

// ---------------------------------------------------------------- conv1x1 in-projection (WMMA), relu
// xi[n, 0..31, p] = relu(W[32,256] @ x[n, 256, p] + b)
__global__ void k_inproj(const float* __restrict__ x, const bf16_t* __restrict__ wbf,
                         const float* __restrict__ bias, float* __restrict__ xi) {
  int lane = threadIdx.x & 31, wv = threadIdx.x >> 5;
  int hi = lane >> 4, l16 = lane & 15;
  int n = blockIdx.y;
  int p0 = blockIdx.x * 64 + wv * 16;
  const float* xb = x + (size_t)n * C_IN * HW;
  fx8 c0 = {0,0,0,0,0,0,0,0}, c1 = {0,0,0,0,0,0,0,0};
  for (int kb = 0; kb < C_IN; kb += 32) {
    bfx16 a0, a1, b;
#pragma unroll
    for (int e = 0; e < 16; ++e) {
      int k = kb + kmap(e, hi);
      a0[e] = wbf[l16 * C_IN + k];
      a1[e] = wbf[(l16 + 16) * C_IN + k];
      b[e]  = (bf16_t)xb[(size_t)k * HW + p0 + l16];
    }
    if (kb + 32 < C_IN)
      __builtin_prefetch(&xb[(size_t)(kb + 32) * HW + p0 + l16], 0, 3);
    c0 = wmma_bf16(a0, b, c0);
    c1 = wmma_bf16(a1, b, c1);
  }
  int col = p0 + l16;
#pragma unroll
  for (int r = 0; r < 8; ++r) {
    int m = r + 8 * hi;
    xi[((size_t)n * NC + m) * HW + col]      = fmaxf(c0[r] + bias[m], 0.f);
    xi[((size_t)n * NC + m + 16) * HW + col] = fmaxf(c1[r] + bias[m + 16], 0.f);
  }
}

// ---------------------------------------------------------------- 3x3 offset conv (WMMA over im2col K=288)
__global__ void k_offconv(const float* __restrict__ xi, const bf16_t* __restrict__ wpad,
                          const float* __restrict__ bias, float* __restrict__ om,
                          int pad, int dil) {
  int lane = threadIdx.x & 31, wv = threadIdx.x >> 5;
  int hi = lane >> 4, l16 = lane & 15;
  int n = blockIdx.y;
  int p0 = blockIdx.x * 64 + wv * 16;
  int pix = p0 + l16, y = pix >> 7, xc = pix & (WW - 1);
  const float* xb = xi + (size_t)n * NC * HW;
  fx8 c0 = {0,0,0,0,0,0,0,0}, c1 = {0,0,0,0,0,0,0,0};
  for (int kb = 0; kb < K_OFF; kb += 32) {
    bfx16 a0, a1, b;
#pragma unroll
    for (int e = 0; e < 16; ++e) {
      int k = kb + kmap(e, hi);
      a0[e] = wpad[l16 * K_OFF + k];
      a1[e] = wpad[(l16 + 16) * K_OFF + k];
      int cch = k / 9, t = k % 9;
      int py = y - pad + (t / 3) * dil;
      int px = xc - pad + (t % 3) * dil;
      float v = 0.f;
      if (py >= 0 && py < HH && px >= 0 && px < WW)
        v = xb[(size_t)cch * HW + py * WW + px];
      b[e] = (bf16_t)v;
    }
    c0 = wmma_bf16(a0, b, c0);
    c1 = wmma_bf16(a1, b, c1);
  }
#pragma unroll
  for (int r = 0; r < 8; ++r) {
    int m = r + 8 * hi;
    if (m < OFF_CH)      om[((size_t)n * OFF_CH + m) * HW + pix]      = c0[r] + bias[m];
    if (m + 16 < OFF_CH) om[((size_t)n * OFF_CH + m + 16) * HW + pix] = c1[r] + bias[m + 16];
  }
}

// ---------------------------------------------------------------- modulated deformable sampling + contraction
__global__ void k_deform(const float* __restrict__ xi, const float* __restrict__ om,
                         const float* __restrict__ w, const float* __restrict__ bias,
                         float* __restrict__ out, int pad, int dil) {
  __shared__ float sw[TR * NC * 9];
  int t = threadIdx.x, n = blockIdx.y;
  for (int i = t; i < TR * NC * 9; i += 256) sw[i] = w[i];
  __syncthreads();
  int pix = blockIdx.x * 256 + t, y = pix >> 7, xc = pix & (WW - 1);
  const float* xb = xi + (size_t)n * NC * HW;
  const float* ob = om + (size_t)n * OFF_CH * HW + pix;
  float acc[TR];
#pragma unroll
  for (int o = 0; o < TR; ++o) acc[o] = 0.f;
  for (int k = 0; k < 9; ++k) {
    float dy = ob[(size_t)(2 * k) * HW];
    float dx = ob[(size_t)(2 * k + 1) * HW];
    float mk = sigmoidf_(ob[(size_t)(18 + k) * HW]);
    float py = (float)(y - pad + (k / 3) * dil) + dy;
    float px = (float)(xc - pad + (k % 3) * dil) + dx;
    float y0 = floorf(py), x0 = floorf(px);
    float cw[4]; int ci[4]; int q = 0;
#pragma unroll
    for (int oy = 0; oy < 2; ++oy)
#pragma unroll
      for (int ox = 0; ox < 2; ++ox) {
        float yi = y0 + oy, xj = x0 + ox;
        float wgt = (1.f - fabsf(py - yi)) * (1.f - fabsf(px - xj));
        bool valid = (yi >= 0.f) && (yi <= (float)(HH - 1)) && (xj >= 0.f) && (xj <= (float)(WW - 1));
        cw[q] = valid ? wgt : 0.f;
        int yc = (int)fminf(fmaxf(yi, 0.f), (float)(HH - 1));
        int xcl = (int)fminf(fmaxf(xj, 0.f), (float)(WW - 1));
        ci[q] = yc * WW + xcl; ++q;
      }
    for (int c = 0; c < NC; ++c) {
      const float* xp = xb + (size_t)c * HW;
      float s = cw[0] * xp[ci[0]] + cw[1] * xp[ci[1]] + cw[2] * xp[ci[2]] + cw[3] * xp[ci[3]];
      s *= mk;
#pragma unroll
      for (int o = 0; o < TR; ++o) acc[o] += sw[o * (NC * 9) + c * 9 + k] * s;
    }
  }
#pragma unroll
  for (int o = 0; o < TR; ++o) out[((size_t)n * TR + o) * HW + pix] = acc[o] + bias[o];
}

// ---------------------------------------------------------------- gk_mask = conv1x1(relu(gkv), gk_w)
__global__ void k_gkmask(const float* __restrict__ gkv, const float* __restrict__ gw,
                         const float* __restrict__ gb, float* __restrict__ gmask) {
  int n = blockIdx.y;
  int pix = blockIdx.x * 256 + threadIdx.x;
  float r[TR];
#pragma unroll
  for (int c = 0; c < TR; ++c) r[c] = fmaxf(gkv[((size_t)n * TR + c) * HW + pix], 0.f);
#pragma unroll
  for (int j = 0; j < NK; ++j) {
    float a = gb[j];
#pragma unroll
    for (int c = 0; c < TR; ++c) a += gw[j * TR + c] * r[c];
    gmask[((size_t)n * NK + j) * HW + pix] = a;
  }
}

// ---------------------------------------------------------------- per-(n,k) spatial max + first argmax
__global__ void k_argmax(const float* __restrict__ gmask, float* __restrict__ maxv,
                         int* __restrict__ maxi) {
  __shared__ float sv[256];
  __shared__ int   si[256];
  int t = threadIdx.x;
  const float* g = gmask + (size_t)blockIdx.x * HW;
  float bv = -3.4e38f; int bi = 0;
  for (int i = t; i < HW; i += 256) { float v = g[i]; if (v > bv) { bv = v; bi = i; } }
  sv[t] = bv; si[t] = bi; __syncthreads();
  for (int s = 128; s > 0; s >>= 1) {
    if (t < s) {
      if (sv[t + s] > sv[t] || (sv[t + s] == sv[t] && si[t + s] < si[t])) {
        sv[t] = sv[t + s]; si[t] = si[t + s];
      }
    }
    __syncthreads();
  }
  if (t == 0) { maxv[blockIdx.x] = sv[0]; maxi[blockIdx.x] = si[0]; }
}

// ---------------------------------------------------------------- keys: gather + pos-embed + gate (tiny)
__global__ void k_key(const float* __restrict__ gkv, const float* __restrict__ maxv,
                      const int* __restrict__ maxi, const float* __restrict__ pos_w,
                      const float* __restrict__ pos_b, float* __restrict__ key) {
  int t = threadIdx.x;
  if (t >= N_B * NK * TR) return;
  int n = t / (NK * TR), rem = t % (NK * TR), k = rem / TR, c = rem % TR;
  int row = k * TR + c;
  float pe = pos_b[row];
#pragma unroll
  for (int j = 0; j < 2 * NK; ++j) {
    int idx = maxi[n * NK + (j % NK)];
    float v = (j < NK) ? (float)(idx / WW) / (float)HH : (float)(idx % WW) / (float)WW;
    pe += pos_w[row * 2 * NK + j] * v;
  }
  int midx = maxi[n * NK + k];
  float g = gkv[((size_t)n * TR + c) * HW + midx];
  key[(n * NK + k) * TR + c] = (g + pe) * sigmoidf_(maxv[n * NK + k]);
}

// ---------------------------------------------------------------- attention (3 keys + self) -> raw xvalue
__global__ void k_attn(const float* __restrict__ xq, const float* __restrict__ gkv,
                       const float* __restrict__ key, float* __restrict__ xval) {
  __shared__ float sk[N_B * NK * TR];
  int t = threadIdx.x, n = blockIdx.y;
  if (t < N_B * NK * TR) sk[t] = key[t];
  __syncthreads();
  int pix = blockIdx.x * 256 + t;
  float q[TR], g[TR];
#pragma unroll
  for (int c = 0; c < TR; ++c) {
    q[c] = xq[((size_t)n * TR + c) * HW + pix];
    g[c] = gkv[((size_t)n * TR + c) * HW + pix];
  }
  const float f = 0.35355339059327373f;  // 1/sqrt(TR)
  float s[NK + 1];
#pragma unroll
  for (int kk = 0; kk < NK; ++kk) {
    float a = 0.f;
#pragma unroll
    for (int c = 0; c < TR; ++c) a += q[c] * sk[n * 24 + c * NK + kk];  // key.reshape(N,TR,NK)
    s[kk] = a * f;
  }
  float a = 0.f;
#pragma unroll
  for (int c = 0; c < TR; ++c) a += q[c] * g[c];
  s[NK] = a * f;
  float m = fmaxf(fmaxf(s[0], s[1]), fmaxf(s[2], s[3]));
  float e0 = expf(s[0] - m), e1 = expf(s[1] - m), e2 = expf(s[2] - m), e3 = expf(s[3] - m);
  float inv = 1.f / (e0 + e1 + e2 + e3);
  e0 *= inv; e1 *= inv; e2 *= inv; e3 *= inv;
#pragma unroll
  for (int c = 0; c < TR; ++c) {
    float xv = e0 * sk[n * 24 + c * NK + 0] + e1 * sk[n * 24 + c * NK + 1] +
               e2 * sk[n * 24 + c * NK + 2] + g[c] * e3;
    xval[((size_t)n * TR + c) * HW + pix] = xv;
  }
}

// ---------------------------------------------------------------- per-(n,c) spatial mean / 1/sqrt(var+eps)
__global__ void k_stats(const float* __restrict__ xval, float* __restrict__ mu,
                        float* __restrict__ rsig) {
  __shared__ float ss[256], sq[256];
  int t = threadIdx.x;
  const float* p = xval + (size_t)blockIdx.x * HW;
  float s = 0.f, q = 0.f;
  for (int i = t; i < HW; i += 256) { float v = p[i]; s += v; q += v * v; }
  ss[t] = s; sq[t] = q; __syncthreads();
  for (int st = 128; st > 0; st >>= 1) {
    if (t < st) { ss[t] += ss[t + st]; sq[t] += sq[t + st]; }
    __syncthreads();
  }
  if (t == 0) {
    float m = ss[0] / (float)HW;
    float var = sq[0] / (float)HW - m * m;
    mu[blockIdx.x] = m;
    rsig[blockIdx.x] = rsqrtf(var + 1e-5f);
  }
}

// ---------------------------------------------------------------- fused relu(concat)+conv1x1+residual (WMMA)
__global__ void k_outproj(const float* __restrict__ xsrc, const float* __restrict__ xq,
                          const float* __restrict__ xval, const float* __restrict__ gkv,
                          const float* __restrict__ gmask, const float* __restrict__ mu,
                          const float* __restrict__ rsig, const bf16_t* __restrict__ wo,
                          const float* __restrict__ ob, float* __restrict__ out) {
  int lane = threadIdx.x & 31, wv = threadIdx.x >> 5;
  int hi = lane >> 4, l16 = lane & 15;
  int n = blockIdx.y;
  int p0 = blockIdx.x * 64 + wv * 16;
  int pix = p0 + l16;
  bfx16 b;
#pragma unroll
  for (int e = 0; e < 16; ++e) {
    int j = kmap(e, hi);
    float v = 0.f;
    if (j < 8) {
      v = xq[((size_t)n * TR + j) * HW + pix];
    } else if (j < 16) {
      int c = j - 8;
      v = (xval[((size_t)n * TR + c) * HW + pix] - mu[n * TR + c]) * rsig[n * TR + c];
    } else if (j < 24) {
      v = gkv[((size_t)n * TR + (j - 16)) * HW + pix];
    } else if (j < 27) {
      v = gmask[((size_t)n * NK + (j - 24)) * HW + pix];
    }
    b[e] = (bf16_t)fmaxf(v, 0.f);
  }
  for (int mt = 0; mt < 16; ++mt) {
    bfx16 a2;
#pragma unroll
    for (int e = 0; e < 16; ++e) a2[e] = wo[(size_t)(mt * 16 + l16) * 32 + kmap(e, hi)];
    fx8 c = {0,0,0,0,0,0,0,0};
    c = wmma_bf16(a2, b, c);
#pragma unroll
    for (int r = 0; r < 8; ++r) {
      int m = mt * 16 + r + 8 * hi;
      size_t o = ((size_t)n * C_IN + m) * HW + pix;
      out[o] = xsrc[o] + ob[m] + c[r];
    }
  }
}

// ---------------------------------------------------------------- host
extern "C" void kernel_launch(void* const* d_in, const int* in_sizes, int n_in,
                              void* d_out, int out_size, void* d_ws, size_t ws_size,
                              hipStream_t stream) {
  (void)in_sizes; (void)n_in; (void)out_size; (void)ws_size;
  const float* x_in = (const float*)d_in[0];
  float* out = (float*)d_out;

  char* base = (char*)d_ws;
  size_t off = 0;
  auto alloc = [&](size_t bytes) -> void* {
    void* p = base + off;
    off += (bytes + 255) & ~(size_t)255;
    return p;
  };
  float*  xi    = (float*)alloc((size_t)N_B * NC * HW * 4);
  float*  om    = (float*)alloc((size_t)N_B * OFF_CH * HW * 4);
  float*  xq    = (float*)alloc((size_t)N_B * TR * HW * 4);
  float*  gkv   = (float*)alloc((size_t)N_B * TR * HW * 4);
  float*  gmask = (float*)alloc((size_t)N_B * NK * HW * 4);
  float*  xval  = (float*)alloc((size_t)N_B * TR * HW * 4);
  bf16_t* w_in  = (bf16_t*)alloc((size_t)NC * C_IN * 2);
  bf16_t* w_q   = (bf16_t*)alloc((size_t)32 * K_OFF * 2);
  bf16_t* w_k   = (bf16_t*)alloc((size_t)32 * K_OFF * 2);
  bf16_t* w_o   = (bf16_t*)alloc((size_t)C_IN * 32 * 2);
  float*  maxv  = (float*)alloc(N_B * NK * 4);
  int*    maxi  = (int*)alloc(N_B * NK * 4);
  float*  keyb  = (float*)alloc(N_B * NK * TR * 4);
  float*  mu    = (float*)alloc(N_B * TR * 4);
  float*  rsig  = (float*)alloc(N_B * TR * 4);

  dim3 gStage((NC * C_IN + 2 * 32 * K_OFF + C_IN * 32 + 255) / 256);
  dim3 gGemm(HW / 64, N_B);   // 4 wave32s per block, 16 pixels each
  dim3 gPix(HW / 256, N_B);

  for (int l = 0; l < 3; ++l) {
    const float* const* pp = (const float* const*)(d_in + 1 + l * 16);
    const float* xsrc = (l == 0) ? x_in : (const float*)out;
    int pad = 2 * l + 3;

    k_stage<<<gStage, 256, 0, stream>>>(pp[0], pp[2], pp[6], pp[14], w_in, w_q, w_k, w_o);
    k_inproj<<<gGemm, 128, 0, stream>>>(xsrc, w_in, pp[1], xi);

    k_offconv<<<gGemm, 128, 0, stream>>>(xi, w_q, pp[3], om, pad, pad);
    k_deform<<<gPix, 256, 0, stream>>>(xi, om, pp[4], pp[5], xq, pad, pad);

    k_offconv<<<gGemm, 128, 0, stream>>>(xi, w_k, pp[7], om, 1, 1);
    k_deform<<<gPix, 256, 0, stream>>>(xi, om, pp[8], pp[9], gkv, 1, 1);

    k_gkmask<<<gPix, 256, 0, stream>>>(gkv, pp[10], pp[11], gmask);
    k_argmax<<<N_B * NK, 256, 0, stream>>>(gmask, maxv, maxi);
    k_key<<<1, 128, 0, stream>>>(gkv, maxv, maxi, pp[12], pp[13], keyb);
    k_attn<<<gPix, 256, 0, stream>>>(xq, gkv, keyb, xval);
    k_stats<<<N_B * TR, 256, 0, stream>>>(xval, mu, rsig);

    k_outproj<<<gGemm, 128, 0, stream>>>(xsrc, xq, xval, gkv, gmask, mu, rsig,
                                         w_o, pp[15], out);
  }
}
